// NonLocalAttention_8581344657896
// MI455X (gfx1250) — matrix operand order repaired and checked by reading, not compiled, not run on hardware
//
#include <hip/hip_runtime.h>

// ---------------------------------------------------------------------------
// Fused windowed attention for gfx1250 (MI455X), f16 WMMA w/ f32 accumulation.
// One workgroup (8 wave32) per 8x8 window; 4096 windows.
// ---------------------------------------------------------------------------

#define CDIM 256
#define NH   8
#define HD   32

typedef _Float16 v16h __attribute__((ext_vector_type(16)));
typedef _Float16 v8h  __attribute__((ext_vector_type(8)));
typedef float    v8f  __attribute__((ext_vector_type(8)));

#define XS  264   // row stride (halves) for X/Q/K buffers (256 + 8 pad, keeps 16B align)
#define VTS 72    // row stride (halves) for transposed V (64 + 8 pad)
#define PBS 72    // row stride (halves) for per-wave P buffer
#define STS 264   // row stride (floats) for f32 staging

#define LDS_X   0
#define LDS_Q   33792
#define LDS_K   67584
#define LDS_VT  101376
#define LDS_TOTAL 138240

// ws layout (bytes)
#define WS_WQKV  0        // 48*8*32*16 halves = 393216 B
#define WS_WPROJ 393216   // 16*8*32*16 halves = 131072 B
#define WS_BIAS  524288   // 8*4*4*32*8 floats = 131072 B
#define WS_BQS   655360   // 256 floats

__device__ __forceinline__ v8f wmma16(v16h a, v16h b, v8f c) {
  return __builtin_amdgcn_wmma_f32_16x16x32_f16(false, a, false, b, (short)0, c,
                                                false, false);
}

__device__ __forceinline__ v16h frag_cat(const _Float16* p0, const _Float16* p1) {
  v8h lo = *(const v8h*)p0;
  v8h hi = *(const v8h*)p1;
  return __builtin_shufflevector(lo, hi, 0,1,2,3,4,5,6,7,8,9,10,11,12,13,14,15);
}

// A fragment (16 x 32, f16): lane half H holds K = 16H + {0..15} (contiguous).
__device__ __forceinline__ v16h load_fragA(const _Float16* base, int row,
                                           int stride, int colbase, int half) {
  const _Float16* p = base + row * stride + colbase + half * 16;
  return frag_cat(p, p + 8);
}
// B fragment (32 x 16, f16): lane half H holds K = 16H + {0..15} (contiguous).
__device__ __forceinline__ v16h load_fragB(const _Float16* base, int row,
                                           int stride, int colbase, int half) {
  const _Float16* p = base + row * stride + colbase + half * 16;
  return frag_cat(p, p + 8);
}

// ---------------------------------------------------------------------------
// Prep 1: f32 weights -> f16 fragments in B-matrix lane order (q pre-scaled).
// ---------------------------------------------------------------------------
__global__ void __launch_bounds__(256)
prep_weights(const float* __restrict__ wq, const float* __restrict__ wkv,
             const float* __restrict__ projw,
             _Float16* __restrict__ wqkv_h, _Float16* __restrict__ wproj_h) {
  const float scale = 0.17677669529663687f;  // 32^-0.5 folded into Wq
  int tid  = blockIdx.x * blockDim.x + threadIdx.x;   // grid = 262144 exactly
  if (tid < 48 * 8 * 32 * 16) {
    int e = tid & 15, lane = (tid >> 4) & 31, frag = tid >> 9;
    int half = lane >> 4, c = lane & 15;
    int nt = frag >> 3, ks = frag & 7;
    int kk = ks * 32 + half * 16 + e;
    int n  = nt * 16 + c;
    float v = (n < 256) ? (wq[kk * 256 + n] * scale) : wkv[kk * 512 + (n - 256)];
    wqkv_h[tid] = (_Float16)v;
  } else {
    int t2 = tid - 48 * 8 * 32 * 16;
    int e = t2 & 15, lane = (t2 >> 4) & 31, frag = t2 >> 9;
    int half = lane >> 4, c = lane & 15;
    int nt = frag >> 3, ks = frag & 7;
    int kk = ks * 32 + half * 16 + e;
    int n  = nt * 16 + c;
    wproj_h[t2] = (_Float16)projw[kk * 256 + n];
  }
}

// ---------------------------------------------------------------------------
// Prep 2: relative-position bias expanded into C-layout fragments + scaled bq.
// ---------------------------------------------------------------------------
__global__ void __launch_bounds__(256)
prep_bias(const float* __restrict__ rpt, const float* __restrict__ bq,
          float* __restrict__ bias_frag, float* __restrict__ bq_s) {
  int tid = blockIdx.x * blockDim.x + threadIdx.x;
  if (tid < 32768) {
    int j = tid & 7, lane = (tid >> 3) & 31;
    int nt = (tid >> 8) & 3, mt = (tid >> 10) & 3, h = tid >> 12;
    int q   = mt * 16 + (lane >> 4) * 8 + j;   // C layout: M = j + 8*half
    int key = nt * 16 + (lane & 15);           // C layout: N = lane&15
    int qi = q >> 3, qj = q & 7, ki = key >> 3, kj = key & 7;
    int idx = (qi - ki + 7) * 15 + (qj - kj + 7);
    bias_frag[tid] = rpt[idx * NH + h];
  } else if (tid < 32768 + 256) {
    bq_s[tid - 32768] = bq[tid - 32768] * 0.17677669529663687f;
  }
}

// ---------------------------------------------------------------------------
// Main fused kernel: load+modulate -> QKV GEMM -> attention -> proj -> store
// ---------------------------------------------------------------------------
__global__ void __launch_bounds__(256)
win_attn_kernel(const float* __restrict__ vid, const float* __restrict__ modulator,
                const float* __restrict__ bkv, const float* __restrict__ proj_b,
                const _Float16* __restrict__ wqkv_h,
                const _Float16* __restrict__ wproj_h,
                const float* __restrict__ bias_frag,
                const float* __restrict__ bq_s, float* __restrict__ out) {
  extern __shared__ char smem[];
  _Float16* Xh = (_Float16*)(smem + LDS_X);    // X, later per-wave P buffers
  _Float16* Qh = (_Float16*)(smem + LDS_Q);    // Q, later attention output O
  _Float16* Kh = (_Float16*)(smem + LDS_K);    // K row-major
  _Float16* Vt = (_Float16*)(smem + LDS_VT);   // V transposed [c][token]
  float*    Stg = (float*)(smem + LDS_K);      // f32 staging, aliases K+Vt

  const int w    = threadIdx.x >> 5;
  const int lane = threadIdx.x & 31;
  const int col  = lane & 15;
  const int half = lane >> 4;

  const int wg = blockIdx.x;
  const int wb = wg & 31;
  const int hb = (wg >> 5) & 31;
  const int tt = wg >> 10;

  // -------- Phase 0: gather window, add modulator, convert to f16 ----------
  for (int rr = 0; rr < 8; ++rr) {
    int row = w * 8 + rr;
    int ti = row >> 3, tj = row & 7;
    long gbase = ((long)((tt * 256 + hb * 8 + ti) * 256 + (wb * 8 + tj))) * CDIM +
                 lane * 8;
    const float4* gp = (const float4*)(vid + gbase);
    float4 a0 = gp[0], a1 = gp[1];
    const float* mp = modulator + row * CDIM + lane * 8;
    v8h p;
    p[0] = (_Float16)(a0.x + mp[0]); p[1] = (_Float16)(a0.y + mp[1]);
    p[2] = (_Float16)(a0.z + mp[2]); p[3] = (_Float16)(a0.w + mp[3]);
    p[4] = (_Float16)(a1.x + mp[4]); p[5] = (_Float16)(a1.y + mp[5]);
    p[6] = (_Float16)(a1.z + mp[6]); p[7] = (_Float16)(a1.w + mp[7]);
    *(v8h*)(Xh + row * XS + lane * 8) = p;
  }
  __syncthreads();

  // -------- Phase 1: QKV GEMM  (64x256) @ (256x768), 24 tiles per wave -----
  for (int chunk = 0; chunk < 2; ++chunk) {
    int ntb = w * 6 + chunk * 3;
    v8f acc[3][4];
#pragma unroll
    for (int i = 0; i < 3; ++i) {
      int nt = ntb + i;
      int n  = nt * 16 + col;
      float b = (nt < 16) ? bq_s[n] : bkv[n - 256];
      v8f bb = {b, b, b, b, b, b, b, b};
#pragma unroll
      for (int mt = 0; mt < 4; ++mt) acc[i][mt] = bb;
    }
#pragma unroll
    for (int ks = 0; ks < 8; ++ks) {
      v16h B[3];
#pragma unroll
      for (int i = 0; i < 3; ++i)
        B[i] = *(const v16h*)(wqkv_h + ((((ntb + i) * 8 + ks) * 32 + lane) << 4));
#pragma unroll
      for (int mt = 0; mt < 4; ++mt) {
        v16h A = load_fragA(Xh, mt * 16 + col, XS, ks * 32, half);
#pragma unroll
        for (int i = 0; i < 3; ++i) acc[i][mt] = wmma16(A, B[i], acc[i][mt]);
      }
    }
#pragma unroll
    for (int i = 0; i < 3; ++i) {
      int nt = ntb + i;
#pragma unroll
      for (int mt = 0; mt < 4; ++mt) {
        int r0 = mt * 16 + half * 8;
        if (nt < 32) {  // Q / K row-major (strided b16 stores)
          _Float16* dst = (nt < 16) ? (Qh + nt * 16 + col)
                                    : (Kh + nt * 16 + col - 256);
#pragma unroll
          for (int j = 0; j < 8; ++j) dst[(r0 + j) * XS] = (_Float16)acc[i][mt][j];
        } else {        // V transposed (contiguous b128 store)
          int cv = nt * 16 + col - 512;
          v8h p;
#pragma unroll
          for (int j = 0; j < 8; ++j) p[j] = (_Float16)acc[i][mt][j];
          *(v8h*)(Vt + cv * VTS + r0) = p;
        }
      }
    }
  }
  __syncthreads();

  // -------- Phase 2: attention, wave w owns head w (head-local columns) ----
  {
    const int h = w;
    _Float16* Pb = Xh + w * (16 * PBS);
#pragma unroll
    for (int mt = 0; mt < 4; ++mt) {
      v8f s[4];
#pragma unroll
      for (int nt = 0; nt < 4; ++nt) {  // C initialized with rel-pos bias
        const float* bp = bias_frag + ((((h * 4 + mt) * 4 + nt) * 32 + lane) << 3);
        float4 b0 = *(const float4*)bp;
        float4 b1 = *(const float4*)(bp + 4);
        v8f sv = {b0.x, b0.y, b0.z, b0.w, b1.x, b1.y, b1.z, b1.w};
        s[nt] = sv;
      }
      v16h qa = load_fragA(Qh, mt * 16 + col, XS, h * HD, half);
#pragma unroll
      for (int nt = 0; nt < 4; ++nt) {   // S = (q*scale) @ K^T + bias
        v16h kb = load_fragB(Kh, nt * 16 + col, XS, h * HD, half);
        s[nt] = wmma16(qa, kb, s[nt]);
      }
      // row softmax: rows live in (vgpr j, lane group of 16)
#pragma unroll
      for (int j = 0; j < 8; ++j) {
        float m = fmaxf(fmaxf(s[0][j], s[1][j]), fmaxf(s[2][j], s[3][j]));
        m = fmaxf(m, __shfl_xor(m, 1, 32));
        m = fmaxf(m, __shfl_xor(m, 2, 32));
        m = fmaxf(m, __shfl_xor(m, 4, 32));
        m = fmaxf(m, __shfl_xor(m, 8, 32));
        float e0 = __expf(s[0][j] - m), e1 = __expf(s[1][j] - m);
        float e2 = __expf(s[2][j] - m), e3 = __expf(s[3][j] - m);
        float sum = e0 + e1 + e2 + e3;
        sum += __shfl_xor(sum, 1, 32);
        sum += __shfl_xor(sum, 2, 32);
        sum += __shfl_xor(sum, 4, 32);
        sum += __shfl_xor(sum, 8, 32);
        float r = 1.0f / sum;
        int rl = half * 8 + j;
        Pb[rl * PBS +  0 + col] = (_Float16)(e0 * r);
        Pb[rl * PBS + 16 + col] = (_Float16)(e1 * r);
        Pb[rl * PBS + 32 + col] = (_Float16)(e2 * r);
        Pb[rl * PBS + 48 + col] = (_Float16)(e3 * r);
      }
      // O(16x32) = P(16x64) @ V(64x32), V read transposed
      v8f o0 = {0, 0, 0, 0, 0, 0, 0, 0}, o1 = o0;
#pragma unroll
      for (int kb = 0; kb < 2; ++kb) {
        v16h pa = load_fragA(Pb, col, PBS, kb * 32, half);
        v16h vb0 = load_fragB(Vt, h * HD + col,      VTS, kb * 32, half);
        v16h vb1 = load_fragB(Vt, h * HD + 16 + col, VTS, kb * 32, half);
        o0 = wmma16(pa, vb0, o0);
        o1 = wmma16(pa, vb1, o1);
      }
      int r0 = mt * 16 + half * 8;
      _Float16* d0 = Qh + h * HD + col;        // overwrite dead Q rows
      _Float16* d1 = Qh + h * HD + 16 + col;
#pragma unroll
      for (int j = 0; j < 8; ++j) {
        d0[(r0 + j) * XS] = (_Float16)o0[j];
        d1[(r0 + j) * XS] = (_Float16)o1[j];
      }
    }
  }
  __syncthreads();

  // -------- Phase 3: output projection (64x256) @ (256x256) ----------------
  {
    v8f acc[2][4];
#pragma unroll
    for (int i = 0; i < 2; ++i) {
      float b = proj_b[(w * 2 + i) * 16 + col];
      v8f bb = {b, b, b, b, b, b, b, b};
#pragma unroll
      for (int mt = 0; mt < 4; ++mt) acc[i][mt] = bb;
    }
#pragma unroll
    for (int ks = 0; ks < 8; ++ks) {
      v16h B[2];
#pragma unroll
      for (int i = 0; i < 2; ++i)
        B[i] = *(const v16h*)(wproj_h + ((((w * 2 + i) * 8 + ks) * 32 + lane) << 4));
#pragma unroll
      for (int mt = 0; mt < 4; ++mt) {
        v16h A = load_fragA(Qh, mt * 16 + col, XS, ks * 32, half);
        acc[0][mt] = wmma16(A, B[0], acc[0][mt]);
        acc[1][mt] = wmma16(A, B[1], acc[1][mt]);
      }
    }
#pragma unroll
    for (int i = 0; i < 2; ++i) {
      float* dst = Stg + (w * 2 + i) * 16 + col;
#pragma unroll
      for (int mt = 0; mt < 4; ++mt) {
        int r0 = mt * 16 + half * 8;
#pragma unroll
        for (int j = 0; j < 8; ++j) dst[(r0 + j) * STS] = acc[i][mt][j];
      }
    }
  }
  __syncthreads();

  // -------- Phase 4: coalesced scatter back into (t,h,w,c) -----------------
  for (int rr = 0; rr < 8; ++rr) {
    int row = w * 8 + rr;
    int ti = row >> 3, tj = row & 7;
    long gbase = ((long)((tt * 256 + hb * 8 + ti) * 256 + (wb * 8 + tj))) * CDIM +
                 lane * 8;
    float4* gp = (float4*)(out + gbase);
    const float* sp = Stg + row * STS + lane * 8;
    float4 r0 = {sp[0], sp[1], sp[2], sp[3]};
    float4 r1 = {sp[4], sp[5], sp[6], sp[7]};
    gp[0] = r0;
    gp[1] = r1;
  }
}

// ---------------------------------------------------------------------------
extern "C" void kernel_launch(void* const* d_in, const int* in_sizes, int n_in,
                              void* d_out, int out_size, void* d_ws, size_t ws_size,
                              hipStream_t stream) {
  const float* vid  = (const float*)d_in[0];
  const float* modu = (const float*)d_in[1];
  const float* wq   = (const float*)d_in[2];
  const float* bq   = (const float*)d_in[3];
  const float* wkv  = (const float*)d_in[4];
  const float* bkv  = (const float*)d_in[5];
  const float* pw   = (const float*)d_in[6];
  const float* pb   = (const float*)d_in[7];
  const float* rpt  = (const float*)d_in[8];

  char* ws = (char*)d_ws;  // ~656 KB used
  _Float16* wqkv_h  = (_Float16*)(ws + WS_WQKV);
  _Float16* wproj_h = (_Float16*)(ws + WS_WPROJ);
  float*    bias_f  = (float*)(ws + WS_BIAS);
  float*    bq_s    = (float*)(ws + WS_BQS);

  prep_weights<<<1024, 256, 0, stream>>>(wq, wkv, pw, wqkv_h, wproj_h);
  prep_bias<<<130, 256, 0, stream>>>(rpt, bq, bias_f, bq_s);
  win_attn_kernel<<<4096, 256, LDS_TOTAL, stream>>>(vid, modu, bkv, pb, wqkv_h,
                                                    wproj_h, bias_f, bq_s,
                                                    (float*)d_out);
}
